// Relation_57303453663499
// MI455X (gfx1250) — compile-verified, matrix operand order checked by reading, not run on hardware
//
#include <hip/hip_runtime.h>
#include <hip/hip_bf16.h>
#include <math.h>

// ---------------------------------------------------------------------------
// Problem constants (match reference)
// ---------------------------------------------------------------------------
#define QN 4096   // queries
#define CN 4096   // classes
#define DN 512    // input dim
#define KN 8      // bilinear slices
#define LN 64     // hash length

typedef __bf16 bf16_t;
typedef __attribute__((ext_vector_type(16))) __bf16 v16bf;
typedef __attribute__((ext_vector_type(8)))  __bf16 v8bf;
typedef __attribute__((ext_vector_type(8)))  float  v8f;

// ---------------------------------------------------------------------------
// WMMA bf16 fragment load.
// ISA 16-bit A-matrix 16x32 layout: lane's 16 halves are two contiguous
// K-runs of 8: K = (lane>>4)*8 + {0..7} and K = 16 + (lane>>4)*8 + {0..7},
// row M = lane&15. We store B transposed ([N][Kdim]) so the same loader (and
// the symmetric B layout, N = lane&15) applies => two b128 loads per lane.
// `p` must already point at (tile row 0, k-chunk 0) of the 16x32 sub-block.
// ---------------------------------------------------------------------------
__device__ __forceinline__ v16bf load_frag_bf16(const bf16_t* __restrict__ p, int ld) {
    const int lane = threadIdx.x & 31;
    const int r    = lane & 15;            // M (or N) index within tile
    const int kb   = (lane >> 4) << 3;     // 0 or 8
    const bf16_t* q = p + (size_t)r * ld + kb;
    v8bf lo = *reinterpret_cast<const v8bf*>(q);        // K = kb .. kb+7
    v8bf hi = *reinterpret_cast<const v8bf*>(q + 16);   // K = 16+kb .. 16+kb+7
    v16bf f;
#pragma unroll
    for (int i = 0; i < 8; ++i) { f[i] = lo[i]; f[i + 8] = hi[i]; }
    return f;
}

// C/D fp32 layout: VGPR r holds M = r + 8*(lane>>4), N = lane&15.
__device__ __forceinline__ void cd_index(int& moff, int& n16) {
    const int lane = threadIdx.x & 31;
    moff = (lane >> 4) << 3;
    n16  = lane & 15;
}

// ---------------------------------------------------------------------------
// Elementwise / transpose prep kernels (one-shot, bandwidth trivial)
// ---------------------------------------------------------------------------
__global__ void k_f32_to_bf16(const float* __restrict__ in, bf16_t* __restrict__ out, int n) {
    int i = blockIdx.x * blockDim.x + threadIdx.x;
    if (i < n) out[i] = (bf16_t)in[i];
}

// classT[d][c] = class[c][d]  (bf16), for the P @ class GEMM (contraction = c)
__global__ void k_class_transpose(const float* __restrict__ cls, bf16_t* __restrict__ ct) {
    int i = blockIdx.x * blockDim.x + threadIdx.x;           // i over C*D, read-coalesced
    if (i >= CN * DN) return;
    int c = i / DN, d = i % DN;
    ct[(size_t)d * CN + c] = (bf16_t)cls[i];
}

// Mt[(k*DN + i)][j] = M[i][j][k]  (bf16), so GEMM1 B operand has contiguous j
__global__ void k_build_mt(const float* __restrict__ M, bf16_t* __restrict__ mt) {
    int idx = blockIdx.x * blockDim.x + threadIdx.x;         // idx over (K*D)*D, write-coalesced
    if (idx >= KN * DN * DN) return;
    int j = idx % DN;
    int i = (idx / DN) % DN;
    int k = idx / (DN * DN);
    mt[idx] = (bf16_t)M[((size_t)i * DN + j) * KN + k];
}

// ---------------------------------------------------------------------------
// Generic wave-tiled bf16 WMMA GEMM:  out[M][N] = A[M][Kdim] * B[N][Kdim]^T
// epi 0: store bf16
// epi 1: += addend fp32 (same [M][ldo] shape), store bf16
// epi 2: tanh(x + bias[n]), store fp32
// ---------------------------------------------------------------------------
__global__ void __launch_bounds__(256)
k_wmma_gemm(const bf16_t* __restrict__ A, int lda,
            const bf16_t* __restrict__ B, int ldb,
            int Mtiles, int Ntiles, int Kdim, int epi,
            bf16_t* __restrict__ outb, float* __restrict__ outf, int ldo,
            const float* __restrict__ addend, const float* __restrict__ bias) {
    const int wave = threadIdx.x >> 5;
    const int t    = blockIdx.x * (blockDim.x >> 5) + wave;   // uniform per wave
    if (t >= Mtiles * Ntiles) return;
    const int tm = t / Ntiles, tn = t % Ntiles;

    const bf16_t* Ab = A + (size_t)tm * 16 * lda;
    const bf16_t* Bb = B + (size_t)tn * 16 * ldb;

    v8f acc = {};
    for (int k0 = 0; k0 < Kdim; k0 += 32) {
        v16bf a = load_frag_bf16(Ab + k0, lda);
        v16bf b = load_frag_bf16(Bb + k0, ldb);
        acc = __builtin_amdgcn_wmma_f32_16x16x32_bf16(false, a, false, b,
                                                      (short)0, acc, false, false);
    }

    int moff, n16; cd_index(moff, n16);
    const int n = tn * 16 + n16;
#pragma unroll
    for (int r = 0; r < 8; ++r) {
        const int m = tm * 16 + moff + r;
        const size_t o = (size_t)m * ldo + n;
        if (epi == 0)      outb[o] = (bf16_t)acc[r];
        else if (epi == 1) outb[o] = (bf16_t)(acc[r] + addend[o]);
        else               outf[o] = tanhf(acc[r] + bias[n]);
    }
}

// ---------------------------------------------------------------------------
// Phase 2 (dominant, 137 GFLOP): fused 8x per-k GEMM + ReLU + score reduce.
// other[q][c] = score_b + sum_k relu( sum_i U[q][k*D+i] * class_bf[c][i] ) * score_w[k]
// One wave -> one 16x16 (q,c) tile, 8 fp32 accumulators (one per k).
// ---------------------------------------------------------------------------
__global__ void __launch_bounds__(256)
k_relation(const bf16_t* __restrict__ U,      // [QN][KN*DN]
           const bf16_t* __restrict__ Cbf,    // [CN][DN]
           const float*  __restrict__ score_w,
           const float*  __restrict__ score_b,
           float* __restrict__ other) {       // [QN][CN]
    const int wave = threadIdx.x >> 5;
    const int t    = blockIdx.x * (blockDim.x >> 5) + wave;
    const int Ntiles = CN / 16;
    if (t >= (QN / 16) * Ntiles) return;
    const int tm = t / Ntiles, tn = t % Ntiles;

    const bf16_t* Ub = U   + (size_t)tm * 16 * (KN * DN);
    const bf16_t* Cb = Cbf + (size_t)tn * 16 * DN;

    v8f zero = {};
    v8f acc[KN];
#pragma unroll
    for (int k = 0; k < KN; ++k) acc[k] = zero;

    for (int i0 = 0; i0 < DN; i0 += 32) {
        v16bf b = load_frag_bf16(Cb + i0, DN);          // class tile, shared by all k
#pragma unroll
        for (int k = 0; k < KN; ++k) {
            v16bf a = load_frag_bf16(Ub + k * DN + i0, KN * DN);
            acc[k] = __builtin_amdgcn_wmma_f32_16x16x32_bf16(false, a, false, b,
                                                             (short)0, acc[k], false, false);
        }
    }

    float w[KN];
#pragma unroll
    for (int k = 0; k < KN; ++k) w[k] = score_w[k];
    const float sb = score_b[0];

    int moff, n16; cd_index(moff, n16);
    const int n = tn * 16 + n16;
#pragma unroll
    for (int r = 0; r < 8; ++r) {
        float s = sb;
#pragma unroll
        for (int k = 0; k < KN; ++k) {
            float v = acc[k][r];
            s += (v > 0.0f ? v : 0.0f) * w[k];
        }
        const int m = tm * 16 + moff + r;
        other[(size_t)m * CN + n] = s;
    }
}

// ---------------------------------------------------------------------------
// Phase 3: row softmax over C, emit bf16 probabilities for the next GEMM.
// One 256-thread block per query row.
// ---------------------------------------------------------------------------
__global__ void __launch_bounds__(256)
k_softmax(const float* __restrict__ other, bf16_t* __restrict__ P) {
    __shared__ float red[256];
    const int q   = blockIdx.x;
    const int tid = threadIdx.x;
    const float* row = other + (size_t)q * CN;

    float lmax = -INFINITY;
    for (int c = tid; c < CN; c += 256) lmax = fmaxf(lmax, row[c]);
    red[tid] = lmax; __syncthreads();
    for (int s = 128; s > 0; s >>= 1) {
        if (tid < s) red[tid] = fmaxf(red[tid], red[tid + s]);
        __syncthreads();
    }
    const float m = red[0]; __syncthreads();

    float lsum = 0.0f;
    for (int c = tid; c < CN; c += 256) lsum += __expf(row[c] - m);
    red[tid] = lsum; __syncthreads();
    for (int s = 128; s > 0; s >>= 1) {
        if (tid < s) red[tid] += red[tid + s];
        __syncthreads();
    }
    const float inv = 1.0f / red[0];

    for (int c = tid; c < CN; c += 256)
        P[(size_t)q * CN + c] = (bf16_t)(__expf(row[c] - m) * inv);
}

// ---------------------------------------------------------------------------
// Host-side orchestration
// ---------------------------------------------------------------------------
static inline size_t align256(size_t x) { return (x + 255) & ~(size_t)255; }

extern "C" void kernel_launch(void* const* d_in, const int* in_sizes, int n_in,
                              void* d_out, int out_size, void* d_ws, size_t ws_size,
                              hipStream_t stream) {
    (void)in_sizes; (void)n_in; (void)out_size; (void)ws_size;

    const float* cls     = (const float*)d_in[0];   // [CN][DN]
    const float* qry     = (const float*)d_in[1];   // [QN][DN]
    const float* Min     = (const float*)d_in[2];   // [DN][DN][KN]
    const float* score_w = (const float*)d_in[3];   // [KN]
    const float* score_b = (const float*)d_in[4];   // [1]
    const float* hash_w  = (const float*)d_in[5];   // [LN][DN]
    const float* hash_b  = (const float*)d_in[6];   // [LN]
    float*       out     = (float*)d_out;           // [QN][LN]

    // Workspace carve-out
    char* p = (char*)d_ws;
    bf16_t* Qbf   = (bf16_t*)p; p += align256((size_t)QN * DN * 2);
    bf16_t* Cbf   = (bf16_t*)p; p += align256((size_t)CN * DN * 2);
    bf16_t* CbfT  = (bf16_t*)p; p += align256((size_t)DN * CN * 2);
    bf16_t* Mt    = (bf16_t*)p; p += align256((size_t)KN * DN * DN * 2);
    bf16_t* Wbf   = (bf16_t*)p; p += align256((size_t)LN * DN * 2);
    bf16_t* U     = (bf16_t*)p; p += align256((size_t)QN * KN * DN * 2);
    float*  other = (float*) p; p += align256((size_t)QN * CN * 4);
    bf16_t* P     = (bf16_t*)p; p += align256((size_t)QN * CN * 2);
    bf16_t* feat  = (bf16_t*)p; p += align256((size_t)QN * DN * 2);

    const int TB = 256;
    // ---- Prep: bf16 converts + transposes
    k_f32_to_bf16<<<(QN * DN + TB - 1) / TB, TB, 0, stream>>>(qry, Qbf, QN * DN);
    k_f32_to_bf16<<<(CN * DN + TB - 1) / TB, TB, 0, stream>>>(cls, Cbf, CN * DN);
    k_f32_to_bf16<<<(LN * DN + TB - 1) / TB, TB, 0, stream>>>(hash_w, Wbf, LN * DN);
    k_class_transpose<<<(CN * DN + TB - 1) / TB, TB, 0, stream>>>(cls, CbfT);
    k_build_mt<<<(KN * DN * DN + TB - 1) / TB, TB, 0, stream>>>(Min, Mt);

    // ---- Phase 1: U[QN][KN*DN] = Qbf[QN][DN] * Mt[KN*DN][DN]^T   (bf16 out)
    {
        int Mtiles = QN / 16, Ntiles = (KN * DN) / 16;
        int blocks = (Mtiles * Ntiles) / 8;
        k_wmma_gemm<<<blocks, TB, 0, stream>>>(Qbf, DN, Mt, DN,
                                               Mtiles, Ntiles, DN, /*epi=*/0,
                                               U, nullptr, KN * DN, nullptr, nullptr);
    }

    // ---- Phase 2: fused bilinear + ReLU + score reduce -> other[QN][CN]
    {
        int blocks = ((QN / 16) * (CN / 16)) / 8;
        k_relation<<<blocks, TB, 0, stream>>>(U, Cbf, score_w, score_b, other);
    }

    // ---- Phase 3: softmax over classes -> P (bf16)
    k_softmax<<<QN, TB, 0, stream>>>(other, P);

    // ---- Phase 4: feat = P * class + query   (bf16 out)
    {
        int Mtiles = QN / 16, Ntiles = DN / 16;
        int blocks = (Mtiles * Ntiles) / 8;
        k_wmma_gemm<<<blocks, TB, 0, stream>>>(P, CN, CbfT, CN,
                                               Mtiles, Ntiles, CN, /*epi=*/1,
                                               feat, nullptr, DN, qry, nullptr);
    }

    // ---- Phase 5: out = tanh(feat * hash_w^T + hash_b)   (fp32 out)
    {
        int Mtiles = QN / 16, Ntiles = LN / 16;
        int blocks = (Mtiles * Ntiles) / 8;
        k_wmma_gemm<<<blocks, TB, 0, stream>>>(feat, DN, Wbf, DN,
                                               Mtiles, Ntiles, DN, /*epi=*/2,
                                               nullptr, out, LN, nullptr, hash_b);
    }
}
